// GAT_11038065951174
// MI455X (gfx1250) — matrix operand order repaired and checked by reading, not compiled, not run on hardware
//
#include <hip/hip_runtime.h>

typedef __attribute__((ext_vector_type(16))) _Float16 v16h;
typedef __attribute__((ext_vector_type(8)))  _Float16 v8h;
typedef __attribute__((ext_vector_type(8)))  float    v8f;

#define NNODES 50000
#define NEDGES 800000
#define ETOT   (NEDGES + NNODES)
#define FIN    128
#define HID    32
#define HEADS  4
#define C1     128           // HEADS*HID
#define NCLS   16
#define NGR    64
#define BN_EPS 1e-5f
#define NEG_SLOPE 0.2f
#define ENC_NEG_INF 0x007FFFFFu   // fenc(-inf)

// ---------- order-preserving float <-> u32 (for atomicMax on floats) ----------
__device__ __forceinline__ unsigned fenc(float f) {
  int i = __float_as_int(f);
  return (i >= 0) ? ((unsigned)i + 0x80000000u) : ~(unsigned)i;
}
__device__ __forceinline__ float fdec(unsigned u) {
  return (u >= 0x80000000u) ? __int_as_float((int)(u - 0x80000000u))
                            : __int_as_float((int)~u);
}

// ---------- fills ----------
__global__ void gat_fill_f32(float* p, float v, int n) {
  int i = blockIdx.x * blockDim.x + threadIdx.x;
  if (i < n) p[i] = v;
}
__global__ void gat_fill_u32(unsigned* p, unsigned v, int n) {
  int i = blockIdx.x * blockDim.x + threadIdx.x;
  if (i < n) p[i] = v;
}

// ---------- BatchNorm: column sums / sums-of-squares over [nrows x 128] ----------
__global__ void gat_bn_stats(const float* __restrict__ x, float* sums, float* sq, int nrows) {
  int c = threadIdx.x;                    // 0..127
  float s = 0.f, s2 = 0.f;
  for (int r = blockIdx.x; r < nrows; r += gridDim.x) {
    float v = x[r * FIN + c];
    s += v; s2 += v * v;
  }
  atomicAdd(&sums[c], s);
  atomicAdd(&sq[c], s2);
}
__global__ void gat_bn_finalize(const float* sums, const float* sq,
                                const float* gamma, const float* beta,
                                float* scale, float* shift, float inv_n) {
  int c = threadIdx.x;
  float mean = sums[c] * inv_n;
  float var  = sq[c] * inv_n - mean * mean;
  float rs   = rsqrtf(var + BN_EPS);
  float sc   = gamma[c] * rs;
  scale[c] = sc;
  shift[c] = beta[c] - mean * sc;
}
__global__ void gat_norm_half(const float* __restrict__ x, const float* scale,
                              const float* shift, _Float16* out, int n) {
  int i = blockIdx.x * blockDim.x + threadIdx.x;
  if (i >= n) return;
  int c = i & (FIN - 1);
  out[i] = (_Float16)(x[i] * scale[c] + shift[c]);
}

// ---------- W[K x Nc] (f32 row-major) -> Wt[Nc x K] (f16) ----------
__global__ void gat_wT_half(const float* __restrict__ W, _Float16* out, int K, int Nc) {
  int t = blockIdx.x * blockDim.x + threadIdx.x;
  if (t >= K * Nc) return;
  int n = t / K, k = t - n * K;
  out[n * K + k] = (_Float16)W[k * Nc + n];
}

// ---------- WMMA GEMM: C[M x Nc] = A[M x 128] * B[128 x Nc], B given transposed ----------
// One wave32 per 16x16 output tile; K=128 -> 4x v_wmma_f32_16x16x32_f16.
// A operand (16b 16x32): lane L -> row L%16, K halves {g*8..g*8+7, 16+g*8..+7}, g=L/16.
// B operand (16b 32x16): lane L -> col L%16, K halves {g*8..g*8+7, 16+g*8..+7}, g=L/16.
// C/D (f32 16x16): acc[v] -> row 8*g+v, col L%16.
__global__ void gat_gemm_wmma(const _Float16* __restrict__ A,
                              const _Float16* __restrict__ Bt,
                              float* __restrict__ C, int M, int Nc) {
  int wave = (blockIdx.x * blockDim.x + threadIdx.x) >> 5;
  int lane = threadIdx.x & 31;
  int ntiles = Nc >> 4;
  int mt = wave / ntiles;
  int nt = wave - mt * ntiles;
  if (mt * 16 >= M) return;                 // wave-uniform: EXEC stays all-ones for WMMA
  int g = lane >> 4;
  int r = lane & 15;
  const _Float16* arow = A  + (mt * 16 + r) * FIN;
  const _Float16* brow = Bt + (nt * 16 + r) * FIN;
  v8f acc = {};
#pragma unroll
  for (int k0 = 0; k0 < FIN; k0 += 32) {
    v8h alo = *(const v8h*)(arow + k0 + g * 8);
    v8h ahi = *(const v8h*)(arow + k0 + 16 + g * 8);
    v16h a = __builtin_shufflevector(alo, ahi, 0,1,2,3,4,5,6,7,8,9,10,11,12,13,14,15);
    v8h blo = *(const v8h*)(brow + k0 + g * 8);
    v8h bhi = *(const v8h*)(brow + k0 + 16 + g * 8);
    v16h b = __builtin_shufflevector(blo, bhi, 0,1,2,3,4,5,6,7,8,9,10,11,12,13,14,15);
    acc = __builtin_amdgcn_wmma_f32_16x16x32_f16(false, a, false, b,
                                                 (short)0, acc, false, false);
  }
  float* crow = C + nt * 16 + r;
#pragma unroll
  for (int v = 0; v < 8; ++v)
    crow[(mt * 16 + g * 8 + v) * Nc] = acc[v];
}

// ---------- attention coefficients: as/ad[n,h] = <h[n,h,:], att_{s,d}[h,:]> ----------
__global__ void gat_attn_coeff(const float* __restrict__ h, const float* att_s,
                               const float* att_d, float* as, float* ad,
                               int n, int heads, int chan) {
  int t = blockIdx.x * blockDim.x + threadIdx.x;
  if (t >= n * heads) return;
  int hh = t % heads;
  const float* hp = h + t * chan;            // t = node*heads + hh, rows contiguous
  float s = 0.f, d = 0.f;
  for (int c = 0; c < chan; ++c) {
    float v = hp[c];
    s += v * att_s[hh * chan + c];
    d += v * att_d[hh * chan + c];
  }
  as[t] = s; ad[t] = d;
}

// ---------- edge helpers ----------
__device__ __forceinline__ void edge_nodes(const long long* __restrict__ ei,
                                           int idx, int& s, int& d) {
  if (idx < NEDGES) { s = (int)ei[idx]; d = (int)ei[NEDGES + idx]; }
  else             { s = d = idx - NEDGES; }   // appended self-loops
}
__device__ __forceinline__ float lrelu(float e) { return e > 0.f ? e : NEG_SLOPE * e; }

__global__ void gat_edge_max(const long long* ei, const float* as, const float* ad,
                             unsigned* emax, int heads) {
  int t = blockIdx.x * blockDim.x + threadIdx.x;
  if (t >= ETOT * heads) return;
  int idx = t / heads, hh = t - idx * heads;
  int s, d; edge_nodes(ei, idx, s, d);
  float e = lrelu(as[s * heads + hh] + ad[d * heads + hh]);
  atomicMax(&emax[d * heads + hh], fenc(e));
}

__global__ void gat_edge_denom(const long long* ei, const float* as, const float* ad,
                               const unsigned* emax, float* den, int heads) {
  int t = blockIdx.x * blockDim.x + threadIdx.x;
  if (t >= ETOT * heads) return;
  int idx = t / heads, hh = t - idx * heads;
  int s, d; edge_nodes(ei, idx, s, d);
  int dh = d * heads + hh;
  float e = lrelu(as[s * heads + hh] + ad[dh]);
  atomicAdd(&den[dh], __expf(e - fdec(emax[dh])));
}

// conv1 scatter: one wave per edge, lane handles 4 of 128 channels (head = chan/32)
__global__ void gat_edge_scatter1(const long long* ei, const float* as, const float* ad,
                                  const unsigned* emax, const float* den,
                                  const float* __restrict__ h, float* acc) {
  int wave = (blockIdx.x * blockDim.x + threadIdx.x) >> 5;
  int lane = threadIdx.x & 31;
  if (wave >= ETOT) return;
  int s, d; edge_nodes(ei, wave, s, d);
  int c0 = lane * 4;
  int hh = c0 >> 5;
  int dh = d * HEADS + hh;
  float e = lrelu(as[s * HEADS + hh] + ad[dh]);
  float alpha = __expf(e - fdec(emax[dh])) / den[dh];
  const float* hp = h + s * C1 + c0;
  float* ap = acc + d * C1 + c0;
#pragma unroll
  for (int i = 0; i < 4; ++i) atomicAdd(&ap[i], hp[i] * alpha);
}

// conv2 scatter: 16 lanes per edge, 1 channel each (heads=1, C=16)
__global__ void gat_edge_scatter2(const long long* ei, const float* as, const float* ad,
                                  const unsigned* emax, const float* den,
                                  const float* __restrict__ h, float* acc) {
  int t = blockIdx.x * blockDim.x + threadIdx.x;
  if (t >= ETOT * NCLS) return;
  int idx = t >> 4, c = t & 15;
  int s, d; edge_nodes(ei, idx, s, d);
  float e = lrelu(as[s] + ad[d]);
  float alpha = __expf(e - fdec(emax[d])) / den[d];
  atomicAdd(&acc[d * NCLS + c], h[s * NCLS + c] * alpha);
}

// ---------- epilogues ----------
__global__ void gat_bias_relu(float* acc, const float* bias, int n) {
  int i = blockIdx.x * blockDim.x + threadIdx.x;
  if (i >= n) return;
  float v = acc[i] + bias[i & (C1 - 1)];
  acc[i] = v > 0.f ? v : 0.f;
}

__global__ void gat_bias_gelu_pool(const float* __restrict__ acc, const float* bias,
                                   const long long* __restrict__ batch,
                                   float* sums, float* cnt, int n) {
  int t = blockIdx.x * blockDim.x + threadIdx.x;
  if (t >= n * NCLS) return;
  int node = t >> 4, c = t & 15;
  float v = acc[t] + bias[c];
  float u = 0.7978845608028654f * (v + 0.044715f * v * v * v);   // jax gelu (tanh approx)
  v = 0.5f * v * (1.f + tanhf(u));
  int g = (int)batch[node];
  atomicAdd(&sums[g * NCLS + c], v);
  if (c == 0) atomicAdd(&cnt[g], 1.f);
}

__global__ void gat_logsoftmax(const float* sums, const float* cnt, float* out) {
  int g = threadIdx.x;
  if (g >= NGR) return;
  float inv = 1.f / fmaxf(cnt[g], 1.f);
  float p[NCLS], mx = -3.0e38f;
  for (int c = 0; c < NCLS; ++c) { p[c] = sums[g * NCLS + c] * inv; mx = fmaxf(mx, p[c]); }
  float se = 0.f;
  for (int c = 0; c < NCLS; ++c) se += __expf(p[c] - mx);
  float lse = mx + logf(se);
  for (int c = 0; c < NCLS; ++c) out[g * NCLS + c] = p[c] - lse;
}

// ================= host side =================
static inline int ceil_div(long long a, int b) { return (int)((a + b - 1) / b); }

extern "C" void kernel_launch(void* const* d_in, const int* in_sizes, int n_in,
                              void* d_out, int out_size, void* d_ws, size_t ws_size,
                              hipStream_t stream) {
  const float* x        = (const float*)d_in[0];
  const float* W1       = (const float*)d_in[1];
  const float* att_src1 = (const float*)d_in[2];
  const float* att_dst1 = (const float*)d_in[3];
  const float* bias1    = (const float*)d_in[4];
  const float* W2       = (const float*)d_in[5];
  const float* att_src2 = (const float*)d_in[6];
  const float* att_dst2 = (const float*)d_in[7];
  const float* bias2    = (const float*)d_in[8];
  const float* bn1_g    = (const float*)d_in[9];
  const float* bn1_b    = (const float*)d_in[10];
  const float* bn2_g    = (const float*)d_in[11];
  const float* bn2_b    = (const float*)d_in[12];
  const long long* ei   = (const long long*)d_in[13];  // int64 [2, E]
  const long long* bat  = (const long long*)d_in[14];  // int64 [N]
  float* out            = (float*)d_out;

  char* ws = (char*)d_ws;
  // workspace layout (bytes)
  float*    stats = (float*)(ws + 0);                 // 1024 f
  float*    sum1  = stats +   0, *sq1 = stats + 128, *scale1 = stats + 256, *shift1 = stats + 384;
  float*    sum2  = stats + 512, *sq2 = stats + 640, *scale2 = stats + 768, *shift2 = stats + 896;
  float*    pool  = (float*)(ws + 4096);              // 1024 f
  float*    cnt   = (float*)(ws + 8192);              // 64 f
  _Float16* xn    = (_Float16*)(ws + 12288);          // N*128 f16
  _Float16* w1t   = (_Float16*)(ws + 12812288);       // 128*128 f16 (W1^T)
  _Float16* w2t   = (_Float16*)(ws + 12845056);       // 16*128 f16 (W2^T)
  float*    h1    = (float*)(ws + 12849152);          // N*128 f32
  float*    acc1  = (float*)(ws + 38449152);          // N*128 f32
  float*    as_   = (float*)(ws + 64049152);          // N*4 f32
  float*    ad_   = (float*)(ws + 64849152);          // N*4 f32
  unsigned* emax  = (unsigned*)(ws + 65649152);       // N*4 u32
  float*    den   = (float*)(ws + 66449152);          // N*4 f32
  float*    h2    = (float*)(ws + 67249152);          // N*16 f32
  float*    acc2  = (float*)(ws + 70449152);          // N*16 f32

  const int B = 256;
  // zero accumulators (workspace is not re-poisoned between replays)
  gat_fill_f32<<<ceil_div(1024, B), B, 0, stream>>>(stats, 0.f, 1024);
  gat_fill_f32<<<ceil_div(1024, B), B, 0, stream>>>(pool, 0.f, 1024);
  gat_fill_f32<<<1, 64, 0, stream>>>(cnt, 0.f, 64);

  // weights -> transposed f16 copies
  gat_wT_half<<<ceil_div(128 * 128, B), B, 0, stream>>>(W1, w1t, 128, 128);
  gat_wT_half<<<ceil_div(128 * 16, B), B, 0, stream>>>(W2, w2t, 128, 16);

  // ---- BN1 + GEMM1 (WMMA) ----
  gat_bn_stats<<<512, 128, 0, stream>>>(x, sum1, sq1, NNODES);
  gat_bn_finalize<<<1, 128, 0, stream>>>(sum1, sq1, bn1_g, bn1_b, scale1, shift1, 1.f / NNODES);
  gat_norm_half<<<ceil_div((long long)NNODES * FIN, B), B, 0, stream>>>(x, scale1, shift1, xn, NNODES * FIN);
  gat_gemm_wmma<<<(NNODES / 16) * (C1 / 16) * 32 / B, B, 0, stream>>>(xn, w1t, h1, NNODES, C1);

  // ---- GATConv1 edge softmax + scatter ----
  gat_attn_coeff<<<ceil_div(NNODES * HEADS, B), B, 0, stream>>>(h1, att_src1, att_dst1, as_, ad_, NNODES, HEADS, HID);
  gat_fill_u32<<<ceil_div(NNODES * HEADS, B), B, 0, stream>>>(emax, ENC_NEG_INF, NNODES * HEADS);
  gat_fill_f32<<<ceil_div(NNODES * HEADS, B), B, 0, stream>>>(den, 0.f, NNODES * HEADS);
  gat_fill_f32<<<ceil_div((long long)NNODES * C1, B), B, 0, stream>>>(acc1, 0.f, NNODES * C1);
  gat_edge_max<<<ceil_div((long long)ETOT * HEADS, B), B, 0, stream>>>(ei, as_, ad_, emax, HEADS);
  gat_edge_denom<<<ceil_div((long long)ETOT * HEADS, B), B, 0, stream>>>(ei, as_, ad_, emax, den, HEADS);
  gat_edge_scatter1<<<ceil_div((long long)ETOT * 32, B), B, 0, stream>>>(ei, as_, ad_, emax, den, h1, acc1);
  gat_bias_relu<<<ceil_div((long long)NNODES * C1, B), B, 0, stream>>>(acc1, bias1, NNODES * C1);

  // ---- BN2 + GEMM2 (WMMA) ----
  gat_bn_stats<<<512, 128, 0, stream>>>(acc1, sum2, sq2, NNODES);
  gat_bn_finalize<<<1, 128, 0, stream>>>(sum2, sq2, bn2_g, bn2_b, scale2, shift2, 1.f / NNODES);
  gat_norm_half<<<ceil_div((long long)NNODES * C1, B), B, 0, stream>>>(acc1, scale2, shift2, xn, NNODES * C1);
  gat_gemm_wmma<<<ceil_div((long long)(NNODES / 16) * 32, B), B, 0, stream>>>(xn, w2t, h2, NNODES, NCLS);

  // ---- GATConv2 edge softmax + scatter (heads=1, C=16) ----
  gat_attn_coeff<<<ceil_div(NNODES, B), B, 0, stream>>>(h2, att_src2, att_dst2, as_, ad_, NNODES, 1, NCLS);
  gat_fill_u32<<<ceil_div(NNODES, B), B, 0, stream>>>(emax, ENC_NEG_INF, NNODES);
  gat_fill_f32<<<ceil_div(NNODES, B), B, 0, stream>>>(den, 0.f, NNODES);
  gat_fill_f32<<<ceil_div(NNODES * NCLS, B), B, 0, stream>>>(acc2, 0.f, NNODES * NCLS);
  gat_edge_max<<<ceil_div((long long)ETOT, B), B, 0, stream>>>(ei, as_, ad_, emax, 1);
  gat_edge_denom<<<ceil_div((long long)ETOT, B), B, 0, stream>>>(ei, as_, ad_, emax, den, 1);
  gat_edge_scatter2<<<ceil_div((long long)ETOT * NCLS, B), B, 0, stream>>>(ei, as_, ad_, emax, den, h2, acc2);

  // ---- GELU + mean-pool + log_softmax ----
  gat_bias_gelu_pool<<<ceil_div(NNODES * NCLS, B), B, 0, stream>>>(acc2, bias2, bat, pool, cnt, NNODES);
  gat_logsoftmax<<<1, 64, 0, stream>>>(pool, cnt, out);
}
